// SpikeFP64LookupExp2_29875792511522
// MI455X (gfx1250) — matrix operand order repaired and checked by reading, not compiled, not run on hardware
//
#include <hip/hip_runtime.h>
#include <hip/hip_bf16.h>

// out[b, :] = onehot64(idx[b]) @ table[64,64], computed with
// V_WMMA_F32_16X16X32_F16 (exact: 0/1 operands, single nonzero per dot).

typedef __attribute__((ext_vector_type(16))) _Float16 v16h;
typedef __attribute__((ext_vector_type(8)))  float    v8f;

__global__ __launch_bounds__(256)
void onehot_wmma_lookup_kernel(const float* __restrict__ table,  // [64 entries][64 bits] f32 (0/1)
                               const int*   __restrict__ idxb,   // [batch][6] int (0/1), MSB first
                               float*       __restrict__ out,    // [batch][64] f32
                               int batch)
{
    const int lane      = threadIdx.x & 31;
    const int waveInBlk = threadIdx.x >> 5;
    const int wavesBlk  = blockDim.x >> 5;
    const int gWave     = blockIdx.x * wavesBlk + waveInBlk;
    const int nWaves    = gridDim.x * wavesBlk;
    const int halfSel   = lane >> 4;     // 0: lanes 0-15, 1: lanes 16-31
    const int l16       = lane & 15;

    // ---- B operands: table in WMMA B layout (K x N = 64 entries x 64 bits),
    // split into 4 N-tiles x 2 K-halves. Loaded once per wave, reused in loop.
    // Lane layout chosen here (contiguous k, not AMD's table order): lanes 0-15 hold K=0..15 of column
    // N=lane; lanes 16-31 hold K=16..31 of column N=lane-16.
    v16h Bop[4][2];
#pragma unroll
    for (int t = 0; t < 4; ++t) {
#pragma unroll
        for (int h = 0; h < 2; ++h) {
            const float* src = table + (size_t)(h * 32 + halfSel * 16) * 64 + t * 16 + l16;
            v16h b;
#pragma unroll
            for (int k = 0; k < 16; ++k)
                b[k] = (_Float16)src[(size_t)k * 64];
            Bop[t][h] = b;
        }
    }

    // A layout used here (contiguous k): lanes 0-15 (row M=lane): elems 0..15 = K0..15.
    // Lanes 16-31 (row M=lane-16): elems 0..15 = K16..31 (matches B).
    const int kb0 = halfSel ? 16 : 0;

    const int nGroups = batch >> 4;   // 16 rows per group
    for (int g = gWave; g < nGroups; g += nWaves) {
        const int rowBase = g << 4;
        const int m = l16;            // this lane's A row within the group

        // idx for row m (lanes L and L+16 load the same 6 ints -> broadcast)
        const int* ib = idxb + (size_t)(rowBase + m) * 6;
        int idx = 0;
#pragma unroll
        for (int k = 0; k < 6; ++k)
            idx = (idx << 1) | (ib[k] > 0 ? 1 : 0);   // MSB first, idx in [0,63]

        // One-hot A: Alo covers K=0..31, Ahi covers K=32..63
        v16h Alo, Ahi;
#pragma unroll
        for (int i = 0; i < 16; ++i) {
            const int K = kb0 + i;  // contiguous: K = 16*halfSel + i
            Alo[i] = (idx == K)      ? (_Float16)1.0f : (_Float16)0.0f;
            Ahi[i] = (idx == K + 32) ? (_Float16)1.0f : (_Float16)0.0f;
        }

#pragma unroll
        for (int t = 0; t < 4; ++t) {
            v8f acc = {};
            acc = __builtin_amdgcn_wmma_f32_16x16x32_f16(
                      false, Alo, false, Bop[t][0], (short)0, acc, false, false);
            acc = __builtin_amdgcn_wmma_f32_16x16x32_f16(
                      false, Ahi, false, Bop[t][1], (short)0, acc, false, false);

            // D layout: VGPR r -> row = rowBase + r + 8*halfSel, col = 16t + l16.
            float* op = out + (size_t)(rowBase + (halfSel ? 8 : 0)) * 64 + t * 16 + l16;
#pragma unroll
            for (int r = 0; r < 8; ++r)
                op[(size_t)r * 64] = acc[r];
        }
    }
}

extern "C" void kernel_launch(void* const* d_in, const int* in_sizes, int n_in,
                              void* d_out, int out_size, void* d_ws, size_t ws_size,
                              hipStream_t stream) {
    const float* table = (const float*)d_in[0];   // [64*64] f32
    const int*   idxb  = (const int*)d_in[1];     // [batch*6] int32
    float*       out   = (float*)d_out;           // [batch*64] f32
    const int batch = in_sizes[1] / 6;            // 262144

    // 512 blocks * 8 waves = 4096 waves; grid-stride over batch/16 = 16384 groups
    dim3 grid(512), block(256);
    onehot_wmma_lookup_kernel<<<grid, block, 0, stream>>>(table, idxb, out, batch);
}
